// WMSA_4088808865805
// MI455X (gfx1250) — matrix operand order repaired and checked by reading, not compiled, not run on hardware
//
#include <hip/hip_runtime.h>

// ---------------------------------------------------------------------------
// Shifted-window MSA (replicating the reference's raw .view() semantics),
// fully fused: 1 workgroup = 1 "window" of 64 tokens, 8 wave32.
// f16 WMMA (v_wmma_f32_16x16x32_f16) with f32 accumulation.
// CDNA5 paths: WMMA, global_load_async_to_lds_b128 + s_wait_asynccnt,
// register softmax via 16-lane xor shuffles.
// ---------------------------------------------------------------------------

typedef __attribute__((ext_vector_type(16))) _Float16 v16h;
typedef __attribute__((ext_vector_type(8)))  float    v8f;

#define H_IMG   192
#define W_IMG   192
#define CDIM    64
#define INNER   512
#define NWIN    2304          // 4 * 192 * 3
#define XS_LD   72            // padded stride for X staging (16B-aligned rows)
#define SCALE_Q 0.125f        // 64^-0.5

// LDS layout in _Float16 units
#define XS_OFF  0
#define QS_OFF  (64 * XS_LD)                  // 4608   (also aliases f32 X stage)
#define KS_OFF  (QS_OFF + 64 * INNER)         // 37376  (K^T slabs; later O)
#define VS_OFF  (KS_OFF + 64 * INNER)         // 70144
#define SMEM_HALVES (VS_OFF + 64 * INNER)     // 102912
#define SMEM_BYTES  (SMEM_HALVES * 2)         // 205824 B  (< 320 KB/WGP)

// f16 weight scratch layout in d_ws (halves)
#define WSQ_OFF 0
#define WSK_OFF 32768          // 64*512
#define WSO_OFF 98304          // + 64*1024
#define WS_HALVES 131072       // + 512*64
#define WS_BYTES (WS_HALVES * 2)

union F16x16 { v16h v; _Float16 h[16]; uint4 q[2]; };
union V8     { v8f  v; float    f[8]; };

__device__ __forceinline__ v8f mma(v16h a, v16h b, v8f c) {
    return __builtin_amdgcn_wmma_f32_16x16x32_f16(
        false, a, false, b, (short)0, c, false, false);
}

// A fragment (16x32 f16, MxK) from LDS row-major (stride ld): lane l holds
// row M=l&15; K runs kb..kb+7 and kb+16..kb+23 (kb=8 for upper half-wave)
// -> two contiguous 16-byte ds_load_b128.
__device__ __forceinline__ v16h ldsA(const _Float16* base, int ld,
                                     int m0, int k0, int lane) {
    const int m  = m0 + (lane & 15);
    const int kb = k0 + ((lane >> 4) << 3);
    const _Float16* p = base + m * ld + kb;
    F16x16 f;
    f.q[0] = *(const uint4*)(p);
    f.q[1] = *(const uint4*)(p + 16);
    return f.v;
}

// B fragment (32x16 f16, KxN): lane l supplies column N=l&15; K runs
// kb..kb+7 and kb+16..kb+23 (kb=8 for upper half-wave). base viewed as [k][n], ld = 64.
__device__ __forceinline__ v16h ldsB_rows(const _Float16* base,
                                          int k0, int n0, int lane) {
    const int n  = n0 + (lane & 15);
    const int kb = k0 + ((lane >> 4) << 3);
    F16x16 f;
#pragma unroll
    for (int t = 0; t < 16; ++t)
        f.h[t] = base[(kb + (t & 7) + ((t >> 3) << 4)) * 64 + n];
    return f.v;
}

// B fragment from a global weight matrix [k][n] (L2-resident).
// WT=_Float16: 16 strided 2-B loads (one per k row), no converts. WT=float: 64 B + cvt (fallback).
template <typename WT>
__device__ __forceinline__ v16h gB(const WT* __restrict__ W, int ldw,
                                   int k0, int n0, int lane) {
    const int n  = n0 + (lane & 15);
    const int kb = k0 + ((lane >> 4) << 3);
    F16x16 f;
#pragma unroll
    for (int t = 0; t < 16; ++t)
        f.h[t] = (_Float16)W[(kb + (t & 7) + ((t >> 3) << 4)) * ldw + n];
    return f.v;
}

// Store a 16x16 f32 C tile as f16 into LDS (row-major, stride ld).
// C layout: VGPR p -> (M=p,N=lane) lanes 0-15; (M=p+8,N=lane-16) lanes 16-31.
__device__ __forceinline__ void stC(_Float16* base, int ld,
                                    int m0, int n0, const V8& c, int lane) {
    const int n  = n0 + (lane & 15);
    const int mo = (lane >> 4) << 3;
#pragma unroll
    for (int p = 0; p < 8; ++p)
        base[(m0 + p + mo) * ld + n] = (_Float16)c.f[p];
}

// Store a K C-tile directly in per-head TRANSPOSED layout KT_h[d][j] so the
// sim GEMM's B fragments hit the contiguous fast path. Token row m, channel
// col c map to: h=m>>3, j=((m&7)<<3)|(c>>6), d=c&63 (row-major view algebra).
__device__ __forceinline__ void stC_KT(_Float16* base, int m0, int c0,
                                       const V8& c, int lane) {
    const int cc = c0 + (lane & 15);
    const int mo = (lane >> 4) << 3;
    const int d  = cc & 63, jlo = cc >> 6;
#pragma unroll
    for (int p = 0; p < 8; ++p) {
        const int m = m0 + p + mo;
        base[((m >> 3) << 12) + d * 64 + (((m & 7) << 3) | jlo)] = (_Float16)c.f[p];
    }
}

template <typename WT>
__global__ void __launch_bounds__(256)
wmsa_fused_kernel(const float* __restrict__ x,
                  const WT* __restrict__ Wq,
                  const WT* __restrict__ Wkv,
                  const WT* __restrict__ Wo,
                  const float* __restrict__ bo,
                  const float* __restrict__ pos,
                  float* __restrict__ out) {
    extern __shared__ _Float16 smem[];
    _Float16* Xs = smem + XS_OFF;   // [64][XS_LD] f16 window tile
    _Float16* Qs = smem + QS_OFF;   // [64][512] Q -> later attn (per-head 64x64)
    _Float16* Ks = smem + KS_OFF;   // per-head K^T slabs -> later O (tok x 512)
    _Float16* Vs = smem + VS_OFF;   // [64][512] V

    const int win  = blockIdx.x;
    const int tid  = threadIdx.x;
    const int wv   = tid >> 5;
    const int lane = tid & 31;

    // Window -> image mapping (roll -4,-4 folded in; the inverse roll on store
    // is the same mapping).
    const int bimg = win / (H_IMG * 3);
    const int rem  = win % (H_IMG * 3);
    const int row  = rem / 3;
    const int cb   = rem % 3;
    int sh = row + 4; if (sh >= H_IMG) sh -= H_IMG;
    const int wbase   = cb * 64 + 4;
    const long hwbase = ((long)bimg * H_IMG + sh) * W_IMG;

    if (tid == 0) {
        __builtin_prefetch(Wq, 0, 3);
        __builtin_prefetch(Wkv, 0, 3);
        __builtin_prefetch(Wo, 0, 3);
    }

    // ---- Phase 1: async-DMA the fp32 window tile (16 KB) into LDS ---------
    // Per-lane addressing handles the %192 wraparound. The f32 staging area
    // aliases the Q slab (dead until Phase 2 writes it, barrier in between).
    {
        const unsigned ldsBase = (unsigned)(QS_OFF * 2);   // byte offset
#pragma unroll
        for (int i = 0; i < 4; ++i) {
            const int c   = tid + i * 256;      // 16-byte chunk id, 1024 total
            const int tok = c >> 4;
            const int ch0 = (c & 15) << 2;
            int sw = wbase + tok; if (sw >= W_IMG) sw -= W_IMG;
            const unsigned long long ga =
                (unsigned long long)(const void*)(x + (hwbase + sw) * CDIM + ch0);
            const unsigned la = ldsBase + (unsigned)(c << 4);
            asm volatile("global_load_async_to_lds_b128 %0, %1, off"
                         :: "v"(la), "v"(ga) : "memory");
        }
        asm volatile("s_wait_asynccnt 0x0" ::: "memory");
        __syncthreads();
        const float* Xf32 = reinterpret_cast<const float*>(smem + QS_OFF);
        for (int idx = tid; idx < 64 * CDIM; idx += 256)
            Xs[(idx >> 6) * XS_LD + (idx & 63)] = (_Float16)Xf32[idx];
    }
    __syncthreads();

    // ---- Phase 2: fused QKV GEMM [64x64] x [64x1536] ----------------------
    // Three straight-line tile loops (uniform W/dst per loop -> no EXEC
    // branching). 128 tiles each, K=64 (2 WMMA steps), 16 tiles/wave.
    const v8f VZ = {0.f, 0.f, 0.f, 0.f, 0.f, 0.f, 0.f, 0.f};
    for (int t = wv; t < 128; t += 8) {          // Q
        const int mt = t & 3, col = (t >> 2) << 4;
        V8 acc; acc.v = VZ;
#pragma unroll
        for (int kc = 0; kc < CDIM; kc += 32)
            acc.v = mma(ldsA(Xs, XS_LD, mt * 16, kc, lane),
                        gB(Wq, 512, kc, col, lane), acc.v);
        stC(Qs, INNER, mt * 16, col, acc, lane);
    }
    for (int t = wv; t < 128; t += 8) {          // K, stored transposed/head
        const int mt = t & 3, col = (t >> 2) << 4;
        V8 acc; acc.v = VZ;
#pragma unroll
        for (int kc = 0; kc < CDIM; kc += 32)
            acc.v = mma(ldsA(Xs, XS_LD, mt * 16, kc, lane),
                        gB(Wkv, 1024, kc, col, lane), acc.v);
        stC_KT(Ks, mt * 16, col, acc, lane);
    }
    for (int t = wv; t < 128; t += 8) {          // V
        const int mt = t & 3, col = (t >> 2) << 4;
        V8 acc; acc.v = VZ;
#pragma unroll
        for (int kc = 0; kc < CDIM; kc += 32)
            acc.v = mma(ldsA(Xs, XS_LD, mt * 16, kc, lane),
                        gB(Wkv, 1024, kc, 512 + col, lane), acc.v);
        stC(Vs, INNER, mt * 16, col, acc, lane);
    }
    __syncthreads();

    // Row-major reshape property: head h's 64x64 matrices are the contiguous
    // slabs at +h*4096. One wave per head.
    _Float16* Qh  = Qs + wv * 4096;   // q' [i][d]
    _Float16* KTh = Ks + wv * 4096;   // k'^T [d][j]
    _Float16* Vh  = Vs + wv * 4096;   // v' [j][d]

    // ---- Phase 3: sim = scale*q'k'^T + pos_emb, register softmax ----------
    V8 acc[4][4];
#pragma unroll
    for (int i = 0; i < 4; ++i)
#pragma unroll
        for (int j = 0; j < 4; ++j) acc[i][j].v = VZ;

#pragma unroll
    for (int kc = 0; kc < 64; kc += 32) {
        v16h aq[4];
#pragma unroll
        for (int ti = 0; ti < 4; ++ti) aq[ti] = ldsA(Qh, 64, ti * 16, kc, lane);
#pragma unroll
        for (int tj = 0; tj < 4; ++tj) {
            v16h bk = ldsB_rows(KTh, kc, tj * 16, lane);   // fast path now
#pragma unroll
            for (int ti = 0; ti < 4; ++ti)
                acc[ti][tj].v = mma(aq[ti], bk, acc[ti][tj].v);
        }
    }
    {
        const int n_lo = lane & 15, mo = (lane >> 4) << 3;
        const float* ph = pos + wv * 64 * 64;
#pragma unroll
        for (int ti = 0; ti < 4; ++ti)
#pragma unroll
            for (int tj = 0; tj < 4; ++tj)
#pragma unroll
                for (int p = 0; p < 8; ++p) {
                    const int M = ti * 16 + p + mo, N = tj * 16 + n_lo;
                    acc[ti][tj].f[p] = acc[ti][tj].f[p] * SCALE_Q + ph[M * 64 + N];
                }
    }
    // Each C-layout register slot holds one row within a 16-lane half:
    // xor-shuffle masks {8,4,2,1} reduce both halves' rows simultaneously.
#pragma unroll
    for (int ti = 0; ti < 4; ++ti)
#pragma unroll
        for (int p = 0; p < 8; ++p) {
            float mx = acc[ti][0].f[p];
#pragma unroll
            for (int tj = 1; tj < 4; ++tj) mx = fmaxf(mx, acc[ti][tj].f[p]);
#pragma unroll
            for (int off = 8; off >= 1; off >>= 1)
                mx = fmaxf(mx, __shfl_xor(mx, off, 32));
            float sum = 0.f;
#pragma unroll
            for (int tj = 0; tj < 4; ++tj) {
                float e = __expf(acc[ti][tj].f[p] - mx);
                acc[ti][tj].f[p] = e;
                sum += e;
            }
#pragma unroll
            for (int off = 8; off >= 1; off >>= 1)
                sum += __shfl_xor(sum, off, 32);
            const float inv = 1.f / sum;
#pragma unroll
            for (int tj = 0; tj < 4; ++tj) acc[ti][tj].f[p] *= inv;
        }
#pragma unroll
    for (int ti = 0; ti < 4; ++ti)       // attn overwrites dead q' slab
#pragma unroll
        for (int tj = 0; tj < 4; ++tj)
            stC(Qh, 64, ti * 16, tj * 16, acc[ti][tj], lane);

    // ---- Phase 4: out'_h = attn * v', O overwrites K^T slab ---------------
    V8 oac[4][4];
#pragma unroll
    for (int i = 0; i < 4; ++i)
#pragma unroll
        for (int j = 0; j < 4; ++j) oac[i][j].v = VZ;
#pragma unroll
    for (int kc = 0; kc < 64; kc += 32) {
        v16h aa[4];
#pragma unroll
        for (int ti = 0; ti < 4; ++ti) aa[ti] = ldsA(Qh, 64, ti * 16, kc, lane);
#pragma unroll
        for (int tj = 0; tj < 4; ++tj) {
            v16h bv = ldsB_rows(Vh, kc, tj * 16, lane);
#pragma unroll
            for (int ti = 0; ti < 4; ++ti)
                oac[ti][tj].v = mma(aa[ti], bv, oac[ti][tj].v);
        }
    }
#pragma unroll
    for (int ti = 0; ti < 4; ++ti)
#pragma unroll
        for (int tj = 0; tj < 4; ++tj)
            stC(Ks + wv * 4096, 64, ti * 16, tj * 16, oac[ti][tj], lane);
    __syncthreads();

    // ---- Phase 5: out = O(64x512) @ Wo(512x64) + bo, scatter with +roll ---
    for (int t = wv; t < 16; t += 8) {
        const int mt = t >> 2, nt = t & 3;
        V8 c; c.v = VZ;
#pragma unroll
        for (int kc = 0; kc < INNER; kc += 32)
            c.v = mma(ldsA(Ks, INNER, mt * 16, kc, lane),
                      gB(Wo, CDIM, kc, nt * 16, lane), c.v);
        const int n_lo = lane & 15, mo = (lane >> 4) << 3;
        const int ch = nt * 16 + n_lo;
        const float bias = bo[ch];
#pragma unroll
        for (int p = 0; p < 8; ++p) {
            const int tok = mt * 16 + p + mo;
            int sw = wbase + tok; if (sw >= W_IMG) sw -= W_IMG;
            out[(hwbase + sw) * CDIM + ch] = c.f[p] + bias;
        }
    }
}

// One-shot weight f32 -> f16 conversion into workspace (shared by all blocks,
// re-done every launch for determinism).
__global__ void convert_weights_kernel(const float* __restrict__ Wq,
                                       const float* __restrict__ Wkv,
                                       const float* __restrict__ Wo,
                                       _Float16* __restrict__ wsh) {
    const int i = blockIdx.x * 256 + threadIdx.x;
    if (i < WSK_OFF)            wsh[i] = (_Float16)Wq[i];
    else if (i < WSO_OFF)       wsh[i] = (_Float16)Wkv[i - WSK_OFF];
    else if (i < WS_HALVES)     wsh[i] = (_Float16)Wo[i - WSO_OFF];
}

extern "C" void kernel_launch(void* const* d_in, const int* in_sizes, int n_in,
                              void* d_out, int out_size, void* d_ws, size_t ws_size,
                              hipStream_t stream) {
    const float* x   = (const float*)d_in[0];
    const float* Wq  = (const float*)d_in[1];
    const float* Wkv = (const float*)d_in[2];
    const float* Wo  = (const float*)d_in[3];
    const float* bo  = (const float*)d_in[4];
    const float* pos = (const float*)d_in[5];
    float* out       = (float*)d_out;
    (void)in_sizes; (void)n_in; (void)out_size;

    if (ws_size >= (size_t)WS_BYTES && d_ws) {
        _Float16* wsh = (_Float16*)d_ws;
        convert_weights_kernel<<<dim3(WS_HALVES / 256), dim3(256), 0, stream>>>(
            Wq, Wkv, Wo, wsh);
        hipFuncSetAttribute((const void*)wmsa_fused_kernel<_Float16>,
                            hipFuncAttributeMaxDynamicSharedMemorySize, SMEM_BYTES);
        wmsa_fused_kernel<_Float16><<<dim3(NWIN), dim3(256), SMEM_BYTES, stream>>>(
            x, wsh + WSQ_OFF, wsh + WSK_OFF, wsh + WSO_OFF, bo, pos, out);
    } else {
        hipFuncSetAttribute((const void*)wmsa_fused_kernel<float>,
                            hipFuncAttributeMaxDynamicSharedMemorySize, SMEM_BYTES);
        wmsa_fused_kernel<float><<<dim3(NWIN), dim3(256), SMEM_BYTES, stream>>>(
            x, Wq, Wkv, Wo, bo, pos, out);
    }
}
